// PointNet_11742440587706
// MI455X (gfx1250) — matrix operand order repaired and checked by reading, not compiled, not run on hardware
//
#include <hip/hip_runtime.h>

typedef __attribute__((ext_vector_type(16))) _Float16 v16h;
typedef __attribute__((ext_vector_type(8)))  _Float16 v8h;
typedef __attribute__((ext_vector_type(8)))  float    v8f;
typedef __attribute__((ext_vector_type(4)))  float    v4f;

#define WPB   8            // waves per block (wave32)
#define BLOCK (WPB * 32)
#define TPW   2            // 16-edge tiles per wave
#define HID   32
#define LDS_STRIDE 40      // halves per staging row (80 B => rows stay 16B-aligned)

// Fragment indices in packed f16 weight buffer (each = 32 lanes x v16h):
//  layer1: [0]=W1 kc0 nt0 [1]=W1 kc0 nt1 [2]=W2 nt0* [3]=W2 nt1*
//  layer2: [4]=W3 kc0 nt0 [5]=W3 kc0 nt1 [6]=W3 kc1 nt0 [7]=W3 kc1 nt1
//          [8]=W4 nt0*    [9]=W4 nt1*          (* = K-permuted for staged GEMM2)
#define NFRAG 10

// ---------------------------------------------------------------------------
// K-permutation for GEMM2. Staging position p holds hidden feature
// pi(p) = (p>>1) + (p&1)*16  (so each lane's (col, col+16) pair is adjacent).
// Lo lanes load positions 0..15 straight into A slots (K 0-7, 16-23);
// hi lanes load positions 16..31 (K 8-15, 24-31). Hardware-K -> position:
//   K in [0,8):   p = K          K in [8,16):  p = 16 + (K-8)
//   K in [16,24): p = 8 + (K-16) K in [24,32): p = 24 + (K-24)
// B fragment row for hardware-K = pi(p(K)).
// ---------------------------------------------------------------------------
__device__ __host__ inline int pn_gemm2_row(int K) {
    const int q = K >> 3, i = K & 7;
    const int p = (q == 0) ? i : (q == 1) ? 16 + i : (q == 2) ? 8 + i : 24 + i;
    return (p >> 1) + ((p & 1) ? 16 : 0);
}

__global__ void pn_pack_weights(const float* __restrict__ W1, const float* __restrict__ W2,
                                const float* __restrict__ W3, const float* __restrict__ W4,
                                v16h* __restrict__ frag) {
    const int tid = threadIdx.x;
    if (tid >= NFRAG * 32) return;
    const int fi = tid >> 5, lane = tid & 31;
    const float* W; int Krows, kc, nt; bool perm;
    switch (fi) {
        case 0: W = W1; Krows = 6;  kc = 0; nt = 0; perm = false; break;
        case 1: W = W1; Krows = 6;  kc = 0; nt = 1; perm = false; break;
        case 2: W = W2; Krows = 32; kc = 0; nt = 0; perm = true;  break;
        case 3: W = W2; Krows = 32; kc = 0; nt = 1; perm = true;  break;
        case 4: W = W3; Krows = 35; kc = 0; nt = 0; perm = false; break;
        case 5: W = W3; Krows = 35; kc = 0; nt = 1; perm = false; break;
        case 6: W = W3; Krows = 35; kc = 1; nt = 0; perm = false; break;
        case 7: W = W3; Krows = 35; kc = 1; nt = 1; perm = false; break;
        case 8: W = W4; Krows = 32; kc = 0; nt = 0; perm = true;  break;
        default:W = W4; Krows = 32; kc = 0; nt = 1; perm = true;  break;
    }
    // B layout: lanes 0-15 = column n, slots = K 0-7,16-23 of chunk; lanes 16-31 = K 8-15,24-31
    const int col = nt * 16 + (lane & 15);
    const int kb  = kc * 32 + ((lane >= 16) ? 8 : 0);
    v16h b;
    #pragma unroll
    for (int t = 0; t < 16; ++t) {
        const int K = kb + (t < 8 ? t : t + 8);
        const int r = perm ? pn_gemm2_row(K) : K;
        b[t] = (r < Krows) ? (_Float16)W[r * HID + col] : (_Float16)0.0f;
    }
    frag[fi * 32 + lane] = b;
}

__global__ void pn_fill_zero(float* __restrict__ a, float* __restrict__ b, int n) {
    int i = blockIdx.x * blockDim.x + threadIdx.x;
    const int stride = gridDim.x * blockDim.x;
    for (; i < n; i += stride) { a[i] = 0.0f; b[i] = 0.0f; }
}

// f32 -> f16 copy of layer-1 output (halves layer-2 gather traffic; A-fragments
// then load halves directly with no cvt chain).
__global__ void pn_cvt_f16(const float* __restrict__ src, _Float16* __restrict__ dst, int n4) {
    int i = blockIdx.x * blockDim.x + threadIdx.x;
    const int stride = gridDim.x * blockDim.x;
    for (; i < n4; i += stride) {
        const v4f x = *(const v4f*)(src + 4 * (size_t)i);
        _Float16* d = dst + 4 * (size_t)i;
        d[0] = (_Float16)x[0]; d[1] = (_Float16)x[1];
        d[2] = (_Float16)x[2]; d[3] = (_Float16)x[3];
    }
}

// ---------------------------------------------------------------------------
// One PointNet edge layer; one wave32 handles TPW tiles of 16 edges via
// V_WMMA_F32_16X16X32_F16. agg pre-zeroed; ReLU folded into positive-only
// atomic segment-max. All LDS is per-wave (wave_barrier = compile fence only).
// ---------------------------------------------------------------------------
template<int F>   // 3: layer1 (A from registers) | 32: layer2 (A from f16 h)
__global__ __launch_bounds__(BLOCK)
void pn_edge_layer(const _Float16* __restrict__ hin16, // [N,32] f16 (layer2 only)
                   const float* __restrict__ pos,      // [N,3]
                   const int*   __restrict__ ei,       // [2,E]
                   int E,
                   const v16h*  __restrict__ frag, int fragBase,
                   const float* __restrict__ ba, const float* __restrict__ bb,
                   float* __restrict__ agg)            // [N,32] pre-zeroed
{
    constexpr int IN = F + 3;
    constexpr int KC = (IN + 31) / 32;

    const int  lane = threadIdx.x & 31;
    const int  wave = threadIdx.x >> 5;
    const int  col  = lane & 15;
    const int  row  = lane & 15;
    const bool hi   = (lane >= 16);

    __shared__ alignas(16) _Float16 smid[WPB][16][LDS_STRIDE];
    __shared__ int sdst[WPB][16];

    // ---- weight fragments: one 32B vector load each
    v16h bA[KC][2];
    #pragma unroll
    for (int kc = 0; kc < KC; ++kc) {
        bA[kc][0] = frag[(fragBase + kc * 2 + 0) * 32 + lane];
        bA[kc][1] = frag[(fragBase + kc * 2 + 1) * 32 + lane];
    }
    const v16h bB0 = frag[(fragBase + KC * 2 + 0) * 32 + lane];
    const v16h bB1 = frag[(fragBase + KC * 2 + 1) * 32 + lane];

    const float ba0 = ba[col], ba1 = ba[col + 16];
    const float bb0 = bb[col], bb1 = bb[col + 16];

    const int waveId = blockIdx.x * WPB + wave;

    #pragma unroll
    for (int it = 0; it < TPW; ++it) {
        const int  tile = waveId * TPW + it;
        const int  e    = tile * 16 + row;
        const bool ev   = (e < E);

        // ---- per-edge gather (lanes r and r+16 handle the same edge row)
        int s = 0, dn = 0;
        float ps0 = 0.f, ps1 = 0.f, ps2 = 0.f;
        float rel0 = 0.f, rel1 = 0.f, rel2 = 0.f;
        if (ev) {
            s  = ei[e];
            dn = ei[E + e];
            ps0 = pos[s * 3 + 0]; ps1 = pos[s * 3 + 1]; ps2 = pos[s * 3 + 2];
            rel0 = ps0 - pos[dn * 3 + 0];
            rel1 = ps1 - pos[dn * 3 + 1];
            rel2 = ps2 - pos[dn * 3 + 2];
        }
        if (!hi) sdst[wave][row] = ev ? dn : -1;

        // ---- GEMM1 (natural K order): A slots 0-7 = K kb..kb+7,
        //      slots 8-15 = K kb+16..kb+23 ; kb = kc*32 + (hi?8:0)
        v8f c0 = {};
        v8f c1 = {};
        #pragma unroll
        for (int kc = 0; kc < KC; ++kc) {
            v16h a;
            #pragma unroll
            for (int t = 0; t < 16; ++t) a[t] = (_Float16)0.0f;

            if constexpr (F == 3) {
                if (!hi && ev) {   // IN=6: lo-lane slots 0..5, all from registers
                    a[0] = (_Float16)ps0;  a[1] = (_Float16)ps1;  a[2] = (_Float16)ps2;
                    a[3] = (_Float16)rel0; a[4] = (_Float16)rel1; a[5] = (_Float16)rel2;
                }
            } else {
                if (kc == 0) {     // f16 h row: 2 x b128, halves already in slot order
                    v8h y0 = {}, y1 = {};
                    if (ev) {
                        const _Float16* hb = hin16 + (size_t)s * HID + (hi ? 8 : 0);
                        y0 = *(const v8h*)(hb);
                        y1 = *(const v8h*)(hb + 16);
                    }
                    #pragma unroll
                    for (int t = 0; t < 8; ++t) { a[t] = y0[t]; a[8 + t] = y1[t]; }
                } else {           // K chunk 1: only k=32..34 (rel), lo lanes
                    if (!hi && ev) {
                        a[0] = (_Float16)rel0; a[1] = (_Float16)rel1; a[2] = (_Float16)rel2;
                    }
                }
            }
            c0 = __builtin_amdgcn_wmma_f32_16x16x32_f16(false, a, false, bA[kc][0],
                                                        (short)0, c0, false, false);
            c1 = __builtin_amdgcn_wmma_f32_16x16x32_f16(false, a, false, bA[kc][1],
                                                        (short)0, c1, false, false);
        }

        // ---- bias + ReLU, stage in permuted-K layout: position 2*col holds
        //      feature col, 2*col+1 holds col+16 -> one b32 store per row
        #pragma unroll
        for (int r = 0; r < 8; ++r) {
            const int rr = (hi ? 8 : 0) + r;
            float m0 = c0[r] + ba0; m0 = m0 > 0.f ? m0 : 0.f;
            float m1 = c1[r] + ba1; m1 = m1 > 0.f ? m1 : 0.f;
            union { _Float16 h2[2]; unsigned u; } pk;
            pk.h2[0] = (_Float16)m0; pk.h2[1] = (_Float16)m1;
            *(unsigned*)&smid[wave][rr][2 * col] = pk.u;
        }
        __builtin_amdgcn_wave_barrier();   // fence: DS in-order within wave

        // ---- GEMM2 (permuted K, matched by packed bB fragments):
        //      lo lanes read positions 0..15, hi lanes 16..31 -> 2 x ds_load_b128
        const int pbase = hi ? 16 : 0;
        const v8h lo8 = *(const v8h*)&smid[wave][row][pbase];
        const v8h hi8 = *(const v8h*)&smid[wave][row][pbase + 8];
        v16h a2;
        #pragma unroll
        for (int t = 0; t < 8; ++t) { a2[t] = lo8[t]; a2[8 + t] = hi8[t]; }

        v8f d0 = {};
        v8f d1 = {};
        d0 = __builtin_amdgcn_wmma_f32_16x16x32_f16(false, a2, false, bB0,
                                                    (short)0, d0, false, false);
        d1 = __builtin_amdgcn_wmma_f32_16x16x32_f16(false, a2, false, bB1,
                                                    (short)0, d1, false, false);

        // ---- bias + positive-only atomic segment-max
        #pragma unroll
        for (int r = 0; r < 8; ++r) {
            const int rr = (hi ? 8 : 0) + r;
            const int d  = sdst[wave][rr];
            const float v0 = d0[r] + bb0;
            const float v1 = d1[r] + bb1;
            if (d >= 0) {
                if (v0 > 0.f) atomicMax((int*)&agg[d * HID + col],      __float_as_int(v0));
                if (v1 > 0.f) atomicMax((int*)&agg[d * HID + col + 16], __float_as_int(v1));
            }
        }
        __builtin_amdgcn_wave_barrier();   // fence before smid/sdst reuse
    }
}

// ---------------------------------------------------------------------------
// inputs: pos, edge_index, batch, W1, b1, W2, b2, W3, b3, W4, b4
// ---------------------------------------------------------------------------
extern "C" void kernel_launch(void* const* d_in, const int* in_sizes, int n_in,
                              void* d_out, int out_size, void* d_ws, size_t ws_size,
                              hipStream_t stream) {
    const float* pos = (const float*)d_in[0];
    const int*   ei  = (const int*)  d_in[1];
    const float* W1  = (const float*)d_in[3];
    const float* b1  = (const float*)d_in[4];
    const float* W2  = (const float*)d_in[5];
    const float* b2  = (const float*)d_in[6];
    const float* W3  = (const float*)d_in[7];
    const float* b3  = (const float*)d_in[8];
    const float* W4  = (const float*)d_in[9];
    const float* b4  = (const float*)d_in[10];

    const int N = in_sizes[0] / 3;     // 100,000 nodes
    const int E = in_sizes[1] / 2;     // 1,600,000 edges

    float*    h1   = (float*)d_ws;                                   // [N,32] f32
    _Float16* h16  = (_Float16*)((char*)d_ws + (size_t)N * HID * 4); // [N,32] f16
    v16h*     frag = (v16h*)((char*)d_ws + (size_t)N * HID * 6);     // 10 KB
    float*    out  = (float*)d_out;                                  // [N,32]

    pn_pack_weights<<<1, NFRAG * 32, 0, stream>>>(W1, W2, W3, W4, frag);
    pn_fill_zero<<<512, 256, 0, stream>>>(h1, out, N * HID);

    const int tiles  = (E + 15) / 16;
    const int blocks = (tiles + WPB * TPW - 1) / (WPB * TPW);

    pn_edge_layer<3 ><<<blocks, BLOCK, 0, stream>>>(nullptr, pos, ei, E, frag, 0, b1, b2, h1);
    pn_cvt_f16<<<512, 256, 0, stream>>>(h1, h16, N * HID / 4);
    pn_edge_layer<32><<<blocks, BLOCK, 0, stream>>>(h16, pos, ei, E, frag, 4, b3, b4, out);
}
